// GPT_75763223101716
// MI455X (gfx1250) — matrix operand order repaired and checked by reading, not compiled, not run on hardware
//
#include <hip/hip_runtime.h>
#include <hip/hip_bf16.h>
#include <math.h>

// ---------------------------------------------------------------------------
// GPT-2 small forward on gfx1250 (MI455X). All dense GEMMs run through
// v_wmma_f32_16x16x32_bf16 (bf16 operands, f32 accumulate), with double-
// buffered LDS staging so global loads overlap WMMA issue.
// ---------------------------------------------------------------------------

typedef __bf16 bf16;
typedef __attribute__((ext_vector_type(16))) __bf16 bf16x16;
typedef __attribute__((ext_vector_type(8)))  __bf16 bf16x8;
typedef __attribute__((ext_vector_type(8)))  float  f32x8;

#define BM 128
#define BN 64
#define BK 64
#define LDS_K 72   // 64 + 8 bf16 pad to spread LDS banks

static __device__ __forceinline__ bf16 f2bf(float f) { return (bf16)f; }

// ---------------------------------------------------------------------------
// Tiled WMMA GEMM:  out(MxN) = epilogue(A(bf16 MxK) @ W(f32 KxN) + bias)
// EPI: 0 = +bias, 1 = +bias then exact GELU, 2 = +bias then + resid (f32)
// OUTBF: true -> store bf16, false -> store f32
// Requirements: M % 128 == 0, K % 64 == 0. N edge handled (head GEMM).
// ---------------------------------------------------------------------------
template<int EPI, bool OUTBF>
__global__ __launch_bounds__(256) void gemm_wmma(
    const bf16*  __restrict__ A,
    const float* __restrict__ W,
    const float* __restrict__ bias,
    const float* __restrict__ resid,
    void*        __restrict__ outv,
    int M, int N, int K)
{
    __shared__ bf16 As[2][BM][LDS_K];    // A tile, row-major (m x k)
    __shared__ bf16 Bt[2][BN][LDS_K];    // W tile, transposed (n x k)

    const int tid  = threadIdx.x;
    const int lane = tid & 31;
    const int wave = tid >> 5;        // 0..7
    const int wm   = wave >> 1;       // 0..3  -> 32-row strip
    const int wn   = wave & 1;        // 0..1  -> 32-col strip
    const int bm   = blockIdx.y * BM;
    const int bn   = blockIdx.x * BN;

    f32x8 acc[2][2];
    #pragma unroll
    for (int i = 0; i < 2; ++i)
        #pragma unroll
        for (int j = 0; j < 2; ++j) {
            f32x8 z = {};
            acc[i][j] = z;
        }

    // ---- cooperative-load assignments ----
    // A: 2 threads per row, 32 bf16 (64B) each
    const int aRow = tid >> 1;            // 0..127
    const int aCol = (tid & 1) * 32;      // 0 or 32
    // W: fixed n per thread, 16 consecutive k (strided global, coalesced per k-row)
    const int nIdx = tid & 63;            // 0..63
    const int kGrp = (tid >> 6) * 16;     // 0,16,32,48

    // ---- fragment lane constants (ISA 7.12.2 16-bit layouts) ----
    const int half = lane >> 4;
    const int l15  = lane & 15;
    const int kbA  = half * 8;            // A: lanes 0-15 k0..7/16..23, 16-31 +8
    const int kbB  = half * 16;           // unused; B reads use kbA, same k order as A

    const bool fullN = (bn + BN) <= N;    // block-uniform; only last head block is partial
    const bf16*  aPtr = A + (size_t)(bm + aRow) * K + aCol;
    const float* wPtr = W + (size_t)kGrp * N + (bn + nIdx);

    uint4 aR0, aR1, aR2, aR3;
    float wreg[16];

    auto loadTile = [&](int k0) {
        const uint4* ag = (const uint4*)(aPtr + k0);
        aR0 = ag[0];
        aR1 = ag[1];
        aR2 = ag[2];
        aR3 = ag[3];
        const float* wg = wPtr + (size_t)k0 * N;
        if (fullN) {
            #pragma unroll
            for (int e = 0; e < 16; ++e) wreg[e] = wg[(size_t)e * N];
        } else {
            #pragma unroll
            for (int e = 0; e < 16; ++e) wreg[e] = 0.0f;
            if (bn + nIdx < N) {
                #pragma unroll
                for (int e = 0; e < 16; ++e) wreg[e] = wg[(size_t)e * N];
            }
        }
    };

    auto stageTile = [&](int buf) {
        *(uint4*)&As[buf][aRow][aCol]      = aR0;
        *(uint4*)&As[buf][aRow][aCol + 8]  = aR1;
        *(uint4*)&As[buf][aRow][aCol + 16] = aR2;
        *(uint4*)&As[buf][aRow][aCol + 24] = aR3;
        bf16x8 t0, t1;
        #pragma unroll
        for (int e = 0; e < 8; ++e) { t0[e] = f2bf(wreg[e]); t1[e] = f2bf(wreg[8 + e]); }
        *(bf16x8*)&Bt[buf][nIdx][kGrp]     = t0;
        *(bf16x8*)&Bt[buf][nIdx][kGrp + 8] = t1;
    };

    const int ktiles = K / BK;
    loadTile(0);
    stageTile(0);
    __syncthreads();

    for (int kt = 0; kt < ktiles; ++kt) {
        const int  cur  = kt & 1;
        const bool more = (kt + 1) < ktiles;
        if (more) loadTile((kt + 1) * BK);
        if (kt + 2 < ktiles) {   // CDNA5 global_prefetch_b8 for tile k+2
            __builtin_prefetch(aPtr + (kt + 2) * BK, 0, 1);
            __builtin_prefetch(wPtr + (size_t)(kt + 2) * BK * N, 0, 1);
        }

        // ---- compute from buffer `cur`: 2 k-substeps x (2x2) WMMA ----
        #pragma unroll
        for (int kk = 0; kk < 2; ++kk) {
            const int ko = kk * 32;
            bf16x16 afrag[2], bfrag[2];
            #pragma unroll
            for (int i = 0; i < 2; ++i) {
                int r = wm * 32 + i * 16 + l15;
                #pragma unroll
                for (int e = 0; e < 8; ++e) {
                    afrag[i][e]     = As[cur][r][ko + kbA + e];
                    afrag[i][8 + e] = As[cur][r][ko + 16 + kbA + e];
                }
            }
            #pragma unroll
            for (int j = 0; j < 2; ++j) {
                int c = wn * 32 + j * 16 + l15;
                #pragma unroll
                for (int e = 0; e < 16; ++e)
                    bfrag[j][e] = Bt[cur][c][ko + (e >> 3) * 16 + kbA + (e & 7)];
            }
            #pragma unroll
            for (int i = 0; i < 2; ++i)
                #pragma unroll
                for (int j = 0; j < 2; ++j)
                    acc[i][j] = __builtin_amdgcn_wmma_f32_16x16x32_bf16(
                        false, afrag[i], false, bfrag[j],
                        (short)0, acc[i][j], false, false);
        }

        if (more) stageTile(cur ^ 1);
        __syncthreads();
    }

    // ---- epilogue: C layout = VGPR r holds row r + 8*half, col = lane&15 ----
    #pragma unroll
    for (int i = 0; i < 2; ++i) {
        #pragma unroll
        for (int j = 0; j < 2; ++j) {
            int col = bn + wn * 32 + j * 16 + l15;
            if (col >= N) continue;
            float bv = bias ? bias[col] : 0.0f;
            #pragma unroll
            for (int r = 0; r < 8; ++r) {
                int row = bm + wm * 32 + i * 16 + half * 8 + r;
                float vv = acc[i][j][r] + bv;
                if (EPI == 1) vv = 0.5f * vv * (1.0f + erff(vv * 0.70710678118654752f));
                if (EPI == 2) vv += resid[(size_t)row * N + col];
                if (OUTBF) ((bf16*)outv)[(size_t)row * N + col] = f2bf(vv);
                else       ((float*)outv)[(size_t)row * N + col] = vv;
            }
        }
    }
}

// ---------------------------------------------------------------------------
// Embedding: x[row] = tok_emb[ids[row]] + pos_emb[row % S]
// ---------------------------------------------------------------------------
__global__ __launch_bounds__(256) void embed_kernel(
    const int* __restrict__ ids, const float* __restrict__ tok,
    const float* __restrict__ pos, float* __restrict__ x, int S, int D)
{
    int row = blockIdx.x;
    int s   = row % S;
    int id  = ids[row];
    const float* t = tok + (size_t)id * D;
    const float* p = pos + (size_t)s  * D;
    float* xr = x + (size_t)row * D;
    for (int d = threadIdx.x; d < D; d += blockDim.x) xr[d] = t[d] + p[d];
}

// ---------------------------------------------------------------------------
// LayerNorm over last dim, writes bf16 (feeds WMMA GEMMs). One block per row.
// ---------------------------------------------------------------------------
__global__ __launch_bounds__(256) void layernorm_bf16_kernel(
    const float* __restrict__ x, const float* __restrict__ g,
    const float* __restrict__ b, bf16* __restrict__ out, int D)
{
    int row = blockIdx.x;
    const float* xr = x + (size_t)row * D;
    float s = 0.f, s2 = 0.f;
    for (int d = threadIdx.x; d < D; d += blockDim.x) {
        float v = xr[d]; s += v; s2 += v * v;
    }
    #pragma unroll
    for (int off = 16; off > 0; off >>= 1) {
        s  += __shfl_xor(s,  off, 32);
        s2 += __shfl_xor(s2, off, 32);
    }
    __shared__ float ws[8], ws2[8];
    int lane = threadIdx.x & 31, wave = threadIdx.x >> 5;
    if (lane == 0) { ws[wave] = s; ws2[wave] = s2; }
    __syncthreads();
    if (wave == 0) {
        s  = (lane < 8) ? ws[lane]  : 0.f;
        s2 = (lane < 8) ? ws2[lane] : 0.f;
        #pragma unroll
        for (int off = 16; off > 0; off >>= 1) {
            s  += __shfl_xor(s,  off, 32);
            s2 += __shfl_xor(s2, off, 32);
        }
        if (lane == 0) { ws[0] = s; ws2[0] = s2; }
    }
    __syncthreads();
    float mu  = ws[0] / D;
    float var = ws2[0] / D - mu * mu;
    float inv = rsqrtf(var + 1e-5f);
    for (int d = threadIdx.x; d < D; d += blockDim.x)
        out[(size_t)row * D + d] = f2bf((xr[d] - mu) * inv * g[d] + b[d]);
}

// ---------------------------------------------------------------------------
// Causal attention, DH=64. One wave (32 lanes) per query row; scores in LDS.
// Causal loop bound (j <= s) implements the -inf mask exactly.
// ---------------------------------------------------------------------------
__global__ __launch_bounds__(128) void attention_kernel(
    const bf16* __restrict__ q, const bf16* __restrict__ k,
    const bf16* __restrict__ v, bf16* __restrict__ o,
    int S, int Hn, int Dtot)
{
    __shared__ float probs[4][1024];
    __shared__ float qs[4][64];

    int wave = threadIdx.x >> 5, lane = threadIdx.x & 31;
    int lin = blockIdx.x * 4 + wave;          // (b*H + h)*S + s
    int s   = lin % S;
    int bh  = lin / S;
    int h   = bh % Hn;
    int b   = bh / Hn;
    size_t base = ((size_t)b * S) * Dtot + (size_t)h * 64;

    const bf16* qrow = q + base + (size_t)s * Dtot;
    qs[wave][lane]      = (float)qrow[lane];
    qs[wave][lane + 32] = (float)qrow[lane + 32];
    __syncthreads();

    float mx = -3.0e38f;
    for (int j = lane; j <= s; j += 32) {
        const bf16* kr = k + base + (size_t)j * Dtot;
        float acc = 0.f;
        #pragma unroll 8
        for (int d = 0; d < 64; ++d) acc += qs[wave][d] * (float)kr[d];
        acc *= 0.125f;  // 1/sqrt(DH)
        probs[wave][j] = acc;
        mx = fmaxf(mx, acc);
    }
    #pragma unroll
    for (int off = 16; off > 0; off >>= 1) mx = fmaxf(mx, __shfl_xor(mx, off, 32));

    float sum = 0.f;
    for (int j = lane; j <= s; j += 32) {
        float e = __expf(probs[wave][j] - mx);
        probs[wave][j] = e;
        sum += e;
    }
    #pragma unroll
    for (int off = 16; off > 0; off >>= 1) sum += __shfl_xor(sum, off, 32);
    float invs = 1.0f / sum;

    float a0 = 0.f, a1 = 0.f;
    for (int j = 0; j <= s; ++j) {
        float p = probs[wave][j];
        const bf16* vr = v + base + (size_t)j * Dtot;
        a0 += p * (float)vr[lane];
        a1 += p * (float)vr[lane + 32];
    }
    bf16* orow = o + base + (size_t)s * Dtot;
    orow[lane]      = f2bf(a0 * invs);
    orow[lane + 32] = f2bf(a1 * invs);
}

// ---------------------------------------------------------------------------
// Host-side orchestration (graph-capture safe: stream launches only)
// ---------------------------------------------------------------------------
extern "C" void kernel_launch(void* const* d_in, const int* in_sizes, int n_in,
                              void* d_out, int out_size, void* d_ws, size_t ws_size,
                              hipStream_t stream)
{
    (void)in_sizes; (void)n_in; (void)out_size;

    const int*   ids  = (const int*)  d_in[0];
    const float* tok  = (const float*)d_in[1];
    const float* pos  = (const float*)d_in[2];
    const float* Wq   = (const float*)d_in[3];  const float* bq = (const float*)d_in[4];
    const float* Wk   = (const float*)d_in[5];  const float* bk = (const float*)d_in[6];
    const float* Wv   = (const float*)d_in[7];  const float* bv = (const float*)d_in[8];
    const float* Wo   = (const float*)d_in[9];  const float* bo = (const float*)d_in[10];
    const float* ln1g = (const float*)d_in[11]; const float* ln1b = (const float*)d_in[12];
    const float* ln2g = (const float*)d_in[13]; const float* ln2b = (const float*)d_in[14];
    const float* W1   = (const float*)d_in[15]; const float* b1 = (const float*)d_in[16];
    const float* W2   = (const float*)d_in[17]; const float* b2 = (const float*)d_in[18];
    const float* lnfg = (const float*)d_in[19]; const float* lnfb = (const float*)d_in[20];
    const float* Whead = (const float*)d_in[21];

    const int B = 2, S = 1024, D = 768, L = 12, H = 12, F = 3072, V = 50257;
    const int M = B * S;

    char* ws = (char*)d_ws;
    size_t off = 0;
    auto alloc = [&](size_t bytes) -> void* {
        void* p = ws + off;
        off += (bytes + 255) & ~(size_t)255;
        return p;
    };
    float* x   = (float*)alloc((size_t)M * D * sizeof(float));
    bf16*  hbf = (bf16*) alloc((size_t)M * D * sizeof(bf16));
    bf16*  qb  = (bf16*) alloc((size_t)M * D * sizeof(bf16));
    bf16*  kb  = (bf16*) alloc((size_t)M * D * sizeof(bf16));
    bf16*  vb  = (bf16*) alloc((size_t)M * D * sizeof(bf16));
    bf16*  ob  = (bf16*) alloc((size_t)M * D * sizeof(bf16));
    bf16*  tb  = (bf16*) alloc((size_t)M * F * sizeof(bf16));
    if (off > ws_size) return;

    embed_kernel<<<M, 256, 0, stream>>>(ids, tok, pos, x, S, D);

    dim3 blk(256);
    dim3 gD((D + BN - 1) / BN, M / BM);   // N = 768
    dim3 gF((F + BN - 1) / BN, M / BM);   // N = 3072

    for (int l = 0; l < L; ++l) {
        const size_t wDD = (size_t)l * D * D;
        const size_t wDF = (size_t)l * D * F;

        layernorm_bf16_kernel<<<M, 256, 0, stream>>>(x, ln1g + l * D, ln1b + l * D, hbf, D);

        gemm_wmma<0, true ><<<gD, blk, 0, stream>>>(hbf, Wq + wDD, bq + l * D, nullptr, qb, M, D, D);
        gemm_wmma<0, true ><<<gD, blk, 0, stream>>>(hbf, Wk + wDD, bk + l * D, nullptr, kb, M, D, D);
        gemm_wmma<0, true ><<<gD, blk, 0, stream>>>(hbf, Wv + wDD, bv + l * D, nullptr, vb, M, D, D);

        attention_kernel<<<(B * H * S) / 4, 128, 0, stream>>>(qb, kb, vb, ob, S, H, D);

        gemm_wmma<2, false><<<gD, blk, 0, stream>>>(ob, Wo + wDD, bo + l * D, x, x, M, D, D);

        layernorm_bf16_kernel<<<M, 256, 0, stream>>>(x, ln2g + l * D, ln2b + l * D, hbf, D);

        gemm_wmma<1, true ><<<gF, blk, 0, stream>>>(hbf, W1 + wDF, b1 + l * F, nullptr, tb, M, F, D);
        gemm_wmma<2, false><<<gD, blk, 0, stream>>>(tb,  W2 + wDF, b2 + l * D, x, x, M, D, F);
    }

    layernorm_bf16_kernel<<<M, 256, 0, stream>>>(x, lnfg, lnfb, hbf, D);
    dim3 gH((V + BN - 1) / BN, M / BM);   // N = 50257 (edge-guarded)
    gemm_wmma<0, false><<<gH, blk, 0, stream>>>(hbf, Whead, nullptr, nullptr, d_out, M, V, D);
}
